// BertForAutoRegressionDecoder_53858889891899
// MI455X (gfx1250) — matrix operand order repaired and checked by reading, not compiled, not run on hardware
//
#include <hip/hip_runtime.h>
#include <cstddef>

// ---------- types ----------
typedef __attribute__((ext_vector_type(16))) __bf16        bf16x16;
typedef __attribute__((ext_vector_type(8)))  float         v8f;
typedef __attribute__((ext_vector_type(4)))  unsigned int  u32x4;

union Frag16 {
  bf16x16        v;
  unsigned short u[16];
  u32x4          q[2];
};

__device__ __forceinline__ v8f wmma_bf16(bf16x16 a, bf16x16 b, v8f c) {
  // D = A(16x32 bf16) * B(32x16 bf16) + C(16x16 f32)
  return __builtin_amdgcn_wmma_f32_16x16x32_bf16(
      /*neg_a=*/false, a, /*neg_b=*/false, b,
      /*c_mod=*/(short)0, c, /*reuse_a=*/false, /*reuse_b=*/false);
}

__device__ __forceinline__ unsigned short f2bf(float f) {
  unsigned int u = __float_as_uint(f);
  u += 0x7FFFu + ((u >> 16) & 1u);   // round-to-nearest-even
  return (unsigned short)(u >> 16);
}

// ---------- model constants ----------
#define NB   2
#define SEQ  2048
#define DM   768
#define NH   12
#define HD   64
#define FFD  3072
#define TDQ  (3 * DM)       // 2304
#define WIN  256
#define MROWS (NB * SEQ)    // 4096

// ---------- copy ----------
__global__ __launch_bounds__(256) void k_copy(const float* __restrict__ src,
                                              float* __restrict__ dst, int n) {
  int i = blockIdx.x * 256 + threadIdx.x;
  if (i < n) dst[i] = src[i];
}

// ---------- weight convert + transpose: W(f32, KxN) -> WT(bf16, NxK) ----------
// 64x64 tiles through LDS so both global streams are coalesced.
__global__ __launch_bounds__(256) void k_cvtT(const float* __restrict__ W,
                                              unsigned short* __restrict__ WT,
                                              int K, int N) {
  __shared__ unsigned short t[64][65];  // padded row to dodge bank conflicts
  const int tid = threadIdx.x;
  const int kb = blockIdx.y * 64;
  const int nb = blockIdx.x * 64;
#pragma unroll
  for (int it = 0; it < 16; ++it) {
    int idx = it * 256 + tid;           // 0..4095
    int kk = idx >> 6, nn = idx & 63;
    t[kk][nn] = f2bf(W[(size_t)(kb + kk) * N + nb + nn]);   // coalesced over n
  }
  __syncthreads();
#pragma unroll
  for (int it = 0; it < 16; ++it) {
    int idx = it * 256 + tid;
    int nn = idx >> 6, kk = idx & 63;
    WT[(size_t)(nb + nn) * K + kb + kk] = t[kk][nn];        // coalesced over k
  }
}

// ---------- layernorm ----------
template <bool BFOUT>
__global__ __launch_bounds__(256) void k_ln(const float* __restrict__ x,
                                            const float* __restrict__ gam,
                                            const float* __restrict__ bet,
                                            void* __restrict__ out, int D) {
  __shared__ float red[256];
  const int row = blockIdx.x;
  const int tid = threadIdx.x;
  const float* xr = x + (size_t)row * D;

  float s = 0.f;
  for (int i = tid; i < D; i += 256) s += xr[i];
  red[tid] = s;
  __syncthreads();
  for (int w = 128; w > 0; w >>= 1) {
    if (tid < w) red[tid] += red[tid + w];
    __syncthreads();
  }
  const float mu = red[0] / (float)D;
  __syncthreads();

  float v = 0.f;
  for (int i = tid; i < D; i += 256) { float d = xr[i] - mu; v += d * d; }
  red[tid] = v;
  __syncthreads();
  for (int w = 128; w > 0; w >>= 1) {
    if (tid < w) red[tid] += red[tid + w];
    __syncthreads();
  }
  const float rstd = rsqrtf(red[0] / (float)D + 1e-12f);

  for (int i = tid; i < D; i += 256) {
    float y = (xr[i] - mu) * rstd * gam[i] + bet[i];
    if (BFOUT) ((unsigned short*)out)[(size_t)row * D + i] = f2bf(y);
    else       ((float*)out)[(size_t)row * D + i] = y;
  }
}

// ---------- GEMM: C = A(bf16, MxK) @ W (given as WT bf16, NxK) + bias ----------
// No LDS, no barriers: A-frag and 4 B-frags are contiguous global loads in the
// same contiguous k order (k = 16*half + s) for A and B; each wave computes 16 rows x 64 cols (4 WMMAs/chunk).
// MODE 0: out = bf16(A@W + b)
// MODE 1: xres += A@W + b           (f32 residual in-place)
// MODE 2: out = bf16(gelu(A@W + b)) (exact gelu)
template <int MODE>
__global__ __launch_bounds__(256) void k_gemm(const unsigned short* __restrict__ A,
                                              const unsigned short* __restrict__ WT,
                                              const float* __restrict__ bias,
                                              unsigned short* __restrict__ Obf,
                                              float* __restrict__ Of32,
                                              int K, int N) {
  const int tid  = threadIdx.x;
  const int wave = tid >> 5;        // 0..7 -> row slab
  const int lane = tid & 31;
  const int half = lane >> 4;
  const int l16  = lane & 15;
  const int rowbase = blockIdx.y * 128 + wave * 16;
  const int colbase = blockIdx.x * 64;

  const v8f vzero = {0.f, 0.f, 0.f, 0.f, 0.f, 0.f, 0.f, 0.f};
  v8f acc[4] = {vzero, vzero, vzero, vzero};

  const unsigned short* arow = A + (size_t)(rowbase + l16) * K;
  const unsigned short* wrow0 = WT + (size_t)(colbase + l16) * K;
  const unsigned short* wrow1 = wrow0 + (size_t)16 * K;
  const unsigned short* wrow2 = wrow0 + (size_t)32 * K;
  const unsigned short* wrow3 = wrow0 + (size_t)48 * K;

  for (int kb = 0; kb < K; kb += 32) {
    Frag16 af;
    af.q[0] = *(const u32x4*)(arow + kb + half * 16);
    af.q[1] = *(const u32x4*)(arow + kb + half * 16 + 8);

    Frag16 b0, b1, b2, b3;
    b0.q[0] = *(const u32x4*)(wrow0 + kb + half * 16);
    b0.q[1] = *(const u32x4*)(wrow0 + kb + half * 16 + 8);
    b1.q[0] = *(const u32x4*)(wrow1 + kb + half * 16);
    b1.q[1] = *(const u32x4*)(wrow1 + kb + half * 16 + 8);
    b2.q[0] = *(const u32x4*)(wrow2 + kb + half * 16);
    b2.q[1] = *(const u32x4*)(wrow2 + kb + half * 16 + 8);
    b3.q[0] = *(const u32x4*)(wrow3 + kb + half * 16);
    b3.q[1] = *(const u32x4*)(wrow3 + kb + half * 16 + 8);

    acc[0] = wmma_bf16(af.v, b0.v, acc[0]);
    acc[1] = wmma_bf16(af.v, b1.v, acc[1]);
    acc[2] = wmma_bf16(af.v, b2.v, acc[2]);
    acc[3] = wmma_bf16(af.v, b3.v, acc[3]);
  }

  // epilogue (C layout: VGPR r -> row r + 8*half, col = lane&15)
#pragma unroll
  for (int t = 0; t < 4; ++t) {
#pragma unroll
    for (int r = 0; r < 8; ++r) {
      int row = rowbase + r + half * 8;
      int col = colbase + t * 16 + l16;
      float v = acc[t][r] + bias[col];
      size_t off = (size_t)row * N + col;
      if (MODE == 0) {
        Obf[off] = f2bf(v);
      } else if (MODE == 1) {
        Of32[off] = Of32[off] + v;
      } else {
        float g = 0.5f * v * (1.0f + erff(v * 0.70710678118654752f));
        Obf[off] = f2bf(g);
      }
    }
  }
}

// ---------- sliding-window causal attention (flash style, WMMA) ----------
// qkv: bf16 [4096, 2304] (q | k | v per head), ctx out: bf16 [4096, 768]
__global__ __launch_bounds__(128) void k_attn(const unsigned short* __restrict__ qkv,
                                              const float* __restrict__ amask,
                                              unsigned short* __restrict__ ctxb) {
  __shared__ __align__(16) unsigned short pbuf[4][16 * 32];  // per-wave P scratch

  const int wave = threadIdx.x >> 5;
  const int lane = threadIdx.x & 31;
  const int half = lane >> 4;
  const int l16  = lane & 15;

  const int tile = blockIdx.x * 4 + wave;          // 0..3071
  const int qt   = tile & 127;                     // S/16 = 128
  const int h    = (tile >> 7) % NH;
  const int b    = tile / (128 * NH);
  const int q0   = qt * 16;

  // Q fragments (hd 0..31 and 32..63), contiguous k order (hd = 16*half + s), direct global loads
  Frag16 qf0, qf1;
  {
    const unsigned short* qrow = qkv + (size_t)(b * SEQ + q0 + l16) * TDQ + h * HD;
    qf0.q[0] = *(const u32x4*)(qrow + half * 16);
    qf0.q[1] = *(const u32x4*)(qrow + half * 16 + 8);
    qf1.q[0] = *(const u32x4*)(qrow + 32 + half * 16);
    qf1.q[1] = *(const u32x4*)(qrow + 32 + half * 16 + 8);
  }

  float run_m[8], run_s[8];
  v8f acc[4];
  const v8f vzero = {0.f, 0.f, 0.f, 0.f, 0.f, 0.f, 0.f, 0.f};
#pragma unroll
  for (int r = 0; r < 8; ++r) { run_m[r] = -3.4028235e38f; run_s[r] = 0.f; }
#pragma unroll
  for (int nt = 0; nt < 4; ++nt) acc[nt] = vzero;

  const int imax    = q0 + 15;
  int jlo = q0 - (WIN - 1); if (jlo < 0) jlo = 0;
  const int c_start = jlo >> 5;
  const int c_end   = imax >> 5;
  const float scale = 0.125f;  // 1/sqrt(64)

  for (int phase = 0; phase < 2; ++phase) {
    int cs, ce;
    if (phase == 0) { if (c_start == 0) continue; cs = 0; ce = 0; }  // sink chunk
    else            { cs = c_start; ce = c_end; }

    for (int c = cs; c <= ce; ++c) {
      const int kbase = c * 32;

      // scores: two 16x16 tiles via 2 WMMAs each (hd = 2 x K32)
      v8f sc[2];
#pragma unroll
      for (int t = 0; t < 2; ++t) {
        const unsigned short* krow =
            qkv + (size_t)(b * SEQ + kbase + t * 16 + l16) * TDQ + DM + h * HD;
        Frag16 k0, k1;
        k0.q[0] = *(const u32x4*)(krow + half * 16);
        k0.q[1] = *(const u32x4*)(krow + half * 16 + 8);
        k1.q[0] = *(const u32x4*)(krow + 32 + half * 16);
        k1.q[1] = *(const u32x4*)(krow + 32 + half * 16 + 8);
        v8f s = vzero;
        s = wmma_bf16(qf0.v, k0.v, s);
        s = wmma_bf16(qf1.v, k1.v, s);
        sc[t] = s;
      }

      float am0 = amask[(size_t)b * SEQ + kbase + l16];
      float am1 = amask[(size_t)b * SEQ + kbase + 16 + l16];

      unsigned short* pls = &pbuf[wave][0];
#pragma unroll
      for (int r = 0; r < 8; ++r) {
        const int i  = q0 + r + half * 8;
        const int j0 = kbase + l16;
        const int j1 = j0 + 16;
        const bool ok0 = (j0 <= i) && (j0 == 0 || j0 + (WIN - 1) >= i);
        const bool ok1 = (j1 <= i) && (j1 == 0 || j1 + (WIN - 1) >= i);
        float s0 = ok0 ? sc[0][r] * scale + am0 : -3.4028235e38f;
        float s1 = ok1 ? sc[1][r] * scale + am1 : -3.4028235e38f;

        float lm = fmaxf(s0, s1);
        lm = fmaxf(lm, __shfl_xor(lm, 1, 16));
        lm = fmaxf(lm, __shfl_xor(lm, 2, 16));
        lm = fmaxf(lm, __shfl_xor(lm, 4, 16));
        lm = fmaxf(lm, __shfl_xor(lm, 8, 16));
        float nm = fmaxf(run_m[r], lm);

        float p0 = ok0 ? __expf(s0 - nm) : 0.f;
        float p1 = ok1 ? __expf(s1 - nm) : 0.f;
        float ls = p0 + p1;
        ls += __shfl_xor(ls, 1, 16);
        ls += __shfl_xor(ls, 2, 16);
        ls += __shfl_xor(ls, 4, 16);
        ls += __shfl_xor(ls, 8, 16);

        float corr = __expf(run_m[r] - nm);
        run_s[r] = run_s[r] * corr + ls;
        run_m[r] = nm;
        acc[0][r] *= corr; acc[1][r] *= corr; acc[2][r] *= corr; acc[3][r] *= corr;

        pls[(r + half * 8) * 32 + l16]      = f2bf(p0);
        pls[(r + half * 8) * 32 + 16 + l16] = f2bf(p1);
      }

      // LDS round-trip: C-layout probs -> A-layout fragment (same wave only)
      asm volatile("s_wait_dscnt 0" ::: "memory");
      Frag16 pf;
      {
        const unsigned short* pw = &pbuf[wave][0];
        pf.q[0] = *(const u32x4*)(pw + l16 * 32 + half * 16);
        pf.q[1] = *(const u32x4*)(pw + l16 * 32 + half * 16 + 8);
      }

      // PV: 4 WMMAs (K = 32 keys), V fragments gathered per B-layout
#pragma unroll
      for (int nt = 0; nt < 4; ++nt) {
        Frag16 vf;
#pragma unroll
        for (int e = 0; e < 16; ++e) {
          int key = kbase + half * 16 + e;
          vf.u[e] = qkv[(size_t)(b * SEQ + key) * TDQ + 2 * DM + h * HD + nt * 16 + l16];
        }
        acc[nt] = wmma_bf16(pf.v, vf.v, acc[nt]);
      }
    }
  }

  // write ctx (bf16), normalized by running sum
#pragma unroll
  for (int nt = 0; nt < 4; ++nt)
#pragma unroll
    for (int r = 0; r < 8; ++r) {
      int row = b * SEQ + q0 + r + half * 8;
      int col = h * HD + nt * 16 + l16;
      ctxb[(size_t)row * DM + col] = f2bf(acc[nt][r] / run_s[r]);
    }
}

// ---------- host-side orchestration ----------
extern "C" void kernel_launch(void* const* d_in, const int* in_sizes, int n_in,
                              void* d_out, int out_size, void* d_ws, size_t ws_size,
                              hipStream_t stream) {
  (void)in_sizes; (void)n_in; (void)out_size; (void)ws_size;

  const float* inp  = (const float*)d_in[0];
  const float* am   = (const float*)d_in[1];
  const float* ln1g = (const float*)d_in[2];
  const float* ln1b = (const float*)d_in[3];
  const float* wqkv = (const float*)d_in[4];
  const float* bqkv = (const float*)d_in[5];
  const float* wao  = (const float*)d_in[6];
  const float* bao  = (const float*)d_in[7];
  const float* ln2g = (const float*)d_in[8];
  const float* ln2b = (const float*)d_in[9];
  const float* wfc  = (const float*)d_in[10];
  const float* bfc  = (const float*)d_in[11];
  const float* wpo  = (const float*)d_in[12];
  const float* bpo  = (const float*)d_in[13];
  const float* lnfg = (const float*)d_in[14];
  const float* lnfb = (const float*)d_in[15];

  char* ws = (char*)d_ws;
  size_t o = 0;
  float*          x     = (float*)(ws + o);          o += (size_t)MROWS * DM * 4;
  unsigned short* hb    = (unsigned short*)(ws + o); o += (size_t)MROWS * DM * 2;
  unsigned short* qkvb  = (unsigned short*)(ws + o); o += (size_t)MROWS * TDQ * 2;
  unsigned short* ctxb  = (unsigned short*)(ws + o); o += (size_t)MROWS * DM * 2;
  unsigned short* fb    = (unsigned short*)(ws + o); o += (size_t)MROWS * FFD * 2;
  unsigned short* wqkvT = (unsigned short*)(ws + o); o += (size_t)DM * TDQ * 2;
  unsigned short* waoT  = (unsigned short*)(ws + o); o += (size_t)DM * DM * 2;
  unsigned short* wfcT  = (unsigned short*)(ws + o); o += (size_t)DM * FFD * 2;
  unsigned short* wpoT  = (unsigned short*)(ws + o); o += (size_t)FFD * DM * 2;

  // x = input_embedding
  {
    int n = MROWS * DM;
    k_copy<<<(n + 255) / 256, 256, 0, stream>>>(inp, x, n);
  }

  for (int l = 0; l < 2; ++l) {
    // convert + transpose this layer's weights to bf16 WT[N][K] (once each)
    k_cvtT<<<dim3(TDQ / 64, DM / 64), 256, 0, stream>>>(
        wqkv + (size_t)l * DM * TDQ, wqkvT, DM, TDQ);
    k_cvtT<<<dim3(DM / 64, DM / 64), 256, 0, stream>>>(
        wao + (size_t)l * DM * DM, waoT, DM, DM);
    k_cvtT<<<dim3(FFD / 64, DM / 64), 256, 0, stream>>>(
        wfc + (size_t)l * DM * FFD, wfcT, DM, FFD);
    k_cvtT<<<dim3(DM / 64, FFD / 64), 256, 0, stream>>>(
        wpo + (size_t)l * FFD * DM, wpoT, FFD, DM);

    // h = LN1(x) -> bf16
    k_ln<true><<<MROWS, 256, 0, stream>>>(x, ln1g + l * DM, ln1b + l * DM, hb, DM);
    // qkv = bf16(h @ Wqkv + b)
    k_gemm<0><<<dim3(TDQ / 64, MROWS / 128), 256, 0, stream>>>(
        hb, wqkvT, bqkv + (size_t)l * TDQ, qkvb, nullptr, DM, TDQ);
    // ctx = attention(q,k,v) -> bf16
    k_attn<<<(NB * NH * (SEQ / 16)) / 4, 128, 0, stream>>>(qkvb, am, ctxb);
    // x += ctx @ Wao + b
    k_gemm<1><<<dim3(DM / 64, MROWS / 128), 256, 0, stream>>>(
        ctxb, waoT, bao + (size_t)l * DM, nullptr, x, DM, DM);
    // h = LN2(x) -> bf16
    k_ln<true><<<MROWS, 256, 0, stream>>>(x, ln2g + l * DM, ln2b + l * DM, hb, DM);
    // f = bf16(gelu(h @ Wfc + b))
    k_gemm<2><<<dim3(FFD / 64, MROWS / 128), 256, 0, stream>>>(
        hb, wfcT, bfc + (size_t)l * FFD, fb, nullptr, DM, FFD);
    // x += f @ Wpo + b
    k_gemm<1><<<dim3(DM / 64, MROWS / 128), 256, 0, stream>>>(
        fb, wpoT, bpo + (size_t)l * DM, nullptr, x, FFD, DM);
  }

  // out = LN_f(x) (f32)
  k_ln<false><<<MROWS, 256, 0, stream>>>(x, lnfg, lnfb, d_out, DM);
}
